// MultiHeadAttention_62689342652572
// MI455X (gfx1250) — compile-verified, matrix operand order checked by reading, not run on hardware
//
#include <hip/hip_runtime.h>
#include <hip/hip_bf16.h>

typedef __attribute__((ext_vector_type(16))) _Float16     v16h;
typedef __attribute__((ext_vector_type(8)))  float        v8f;
typedef __attribute__((ext_vector_type(4)))  unsigned int v4u;
typedef __attribute__((ext_vector_type(8)))  int          v8i;
typedef __attribute__((ext_vector_type(4)))  int          v4i;

#define WMMA_F32_F16(a, b, c) \
  __builtin_amdgcn_wmma_f32_16x16x32_f16(false, (a), false, (b), (short)0, (c), false, false)

// element e (0..15) of a 16-bit A/B WMMA fragment -> K index (0..31)
// lane half 0 owns K {0..7, 16..23}, half 1 owns K {8..15, 24..31}
__device__ __forceinline__ int kmap(int e, int half) {
  return (e < 8) ? (half * 8 + e) : (16 + half * 8 + (e - 8));
}

// ---------------------------------------------------------------------------
// TDM: issue a 2D tile load (tile_k x tile_rows, f32) from global to LDS.
// D# layout per CDNA5 ISA 8.3/8.4:
//  group0: [1:0]=count=1, [63:32]=lds_addr, [120:64]=global_addr, [127:126]=type(2)
//  group1: [17:16]=data_size(2 -> 4B), [20]=pad_enable,
//          [24:22]=pad_interval(4 -> 32 DWORDs), [31:25]=pad_amount(3 -> 4 DWORDs),
//          [79:48]=tensor_dim0, [111:80]=tensor_dim1, [127:112]=tile_dim0,
//          [143:128]=tile_dim1, [207:160]=tensor_dim0_stride
//  groups 2/3: zero (2D tile: tile_dim2=0, dims unused)
// LDS row stride becomes 32+4=36 DWORDs (144B, 16B-aligned rows) -> lets the
// fragment reads lower to ds_load_b128 and avoids bank conflicts.
// ---------------------------------------------------------------------------
__device__ __forceinline__ void tdm_load_2d_f32(
    unsigned int lds_off, const float* gptr,
    unsigned int tile_k, unsigned int tile_rows, unsigned int row_stride)
{
  const unsigned long long ga = (unsigned long long)(size_t)gptr;
  v4u g0;
  g0[0] = 1u;                                            // count=1
  g0[1] = lds_off;                                       // lds_addr
  g0[2] = (unsigned int)(ga & 0xFFFFFFFFu);              // global_addr[31:0]
  g0[3] = (unsigned int)((ga >> 32) & 0x1FFFFFFu) | (2u << 30);  // [56:32] | type=2
  v8i g1;
  g1[0] = (int)((2u << 16) | (1u << 20) | (4u << 22) | (3u << 25)); // 4B, pad 4dw/32dw
  g1[1] = (int)((row_stride & 0xFFFFu) << 16);           // tensor_dim0 lo16 (=K)
  g1[2] = (int)(((row_stride >> 16) & 0xFFFFu) | (4096u << 16)); // dim0 hi | dim1 lo
  g1[3] = (int)(((4096u >> 16) & 0xFFFFu) | (tile_k << 16));     // dim1 hi | tile_dim0
  g1[4] = (int)tile_rows;                                // tile_dim1 (tile_dim2=0)
  g1[5] = (int)row_stride;                               // tensor_dim0_stride lo32
  g1[6] = 0;                                             // stride hi | dim1_stride lo
  g1[7] = 0;
  const v4i gz4 = {0, 0, 0, 0};
  const v8i gz8 = {0, 0, 0, 0, 0, 0, 0, 0};
  __builtin_amdgcn_tensor_load_to_lds(g0, g1, gz4, gz4, gz8, 0);
}

// ---------------------------------------------------------------------------
// proj_gemm: Out[4096,1024] = X[4096,1024] @ W[1024,1024]^T + bias
// Workgroup tile 128(M) x 64(N), 8 waves, each 32x32 via 4 wmma tiles.
// TDM double-buffered LDS pipeline (f32 tiles, cvt to f16 on fragment build).
// ---------------------------------------------------------------------------
#define MDIM 4096
#define NDIM 1024
#define KDIM 1024
#define PT_M 128
#define PT_N 64
#define PT_K 32
#define LROW (PT_K + 4)          // 36 floats: pad_amount=4 DWORDs per 32
#define NK   (KDIM / PT_K)       // 32 k-steps

__global__ __launch_bounds__(256) void proj_gemm_kernel(
    const float* __restrict__ X, const float* __restrict__ W,
    const float* __restrict__ bias, float* __restrict__ Out)
{
  __shared__ __align__(16) float As[2][PT_M][LROW];
  __shared__ __align__(16) float Bs[2][PT_N][LROW];

  const int tid  = threadIdx.x;
  const int lane = tid & 31;
  const int wave = tid >> 5;     // 0..7
  const int wm   = wave & 3;     // 4 waves along M
  const int wn   = wave >> 2;    // 2 waves along N
  const int half = lane >> 4;
  const int l16  = lane & 15;
  const int m0   = blockIdx.y * PT_M;
  const int n0   = blockIdx.x * PT_N;

  v8f c[2][2];
#pragma unroll
  for (int mi = 0; mi < 2; ++mi)
#pragma unroll
    for (int ni = 0; ni < 2; ++ni) {
      const float bv = bias[n0 + wn * 32 + ni * 16 + l16];
#pragma unroll
      for (int r = 0; r < 8; ++r) c[mi][ni][r] = bv;
    }

  const unsigned int asOff[2] = {(unsigned int)(size_t)&As[0][0][0],
                                 (unsigned int)(size_t)&As[1][0][0]};
  const unsigned int bsOff[2] = {(unsigned int)(size_t)&Bs[0][0][0],
                                 (unsigned int)(size_t)&Bs[1][0][0]};

  // prologue: wave 0 issues DMA for k-steps 0 and 1
  if (tid < 32) {
    tdm_load_2d_f32(asOff[0], X + (size_t)m0 * KDIM + 0 * PT_K, PT_K, PT_M, KDIM);
    tdm_load_2d_f32(bsOff[0], W + (size_t)n0 * KDIM + 0 * PT_K, PT_K, PT_N, KDIM);
    tdm_load_2d_f32(asOff[1], X + (size_t)m0 * KDIM + 1 * PT_K, PT_K, PT_M, KDIM);
    tdm_load_2d_f32(bsOff[1], W + (size_t)n0 * KDIM + 1 * PT_K, PT_K, PT_N, KDIM);
  }

  for (int i = 0; i < NK; ++i) {
    const int buf = i & 1;
    if (tid < 32) {
      if (i + 1 < NK) __builtin_amdgcn_s_wait_tensorcnt(2);  // DMA(i) retired
      else            __builtin_amdgcn_s_wait_tensorcnt(0);
    }
    __syncthreads();             // publish buf to all waves

    v16h a[2], bfr[2];
#pragma unroll
    for (int mi = 0; mi < 2; ++mi) {
      const int row = wm * 32 + mi * 16 + l16;
#pragma unroll
      for (int e = 0; e < 16; ++e)
        a[mi][e] = (_Float16)As[buf][row][kmap(e, half)];
    }
#pragma unroll
    for (int ni = 0; ni < 2; ++ni) {
      const int col = wn * 32 + ni * 16 + l16;
#pragma unroll
      for (int e = 0; e < 16; ++e)
        bfr[ni][e] = (_Float16)Bs[buf][col][kmap(e, half)];
    }
#pragma unroll
    for (int mi = 0; mi < 2; ++mi)
#pragma unroll
      for (int ni = 0; ni < 2; ++ni)
        c[mi][ni] = WMMA_F32_F16(a[mi], bfr[ni], c[mi][ni]);

    __syncthreads();             // all waves done reading buf
    if (tid < 32 && i + 2 < NK) {
      tdm_load_2d_f32(asOff[buf], X + (size_t)m0 * KDIM + (size_t)(i + 2) * PT_K,
                      PT_K, PT_M, KDIM);
      tdm_load_2d_f32(bsOff[buf], W + (size_t)n0 * KDIM + (size_t)(i + 2) * PT_K,
                      PT_K, PT_N, KDIM);
    }
  }

#pragma unroll
  for (int mi = 0; mi < 2; ++mi)
#pragma unroll
    for (int ni = 0; ni < 2; ++ni) {
      const int gcol = n0 + wn * 32 + ni * 16 + l16;
#pragma unroll
      for (int r = 0; r < 8; ++r) {
        const int grow = m0 + wm * 32 + mi * 16 + r + 8 * half;
        Out[(size_t)grow * NDIM + gcol] = c[mi][ni][r];
      }
    }
}

// ---------------------------------------------------------------------------
// attn_scores: one wave per (b, h, 16-row q tile).
// Raw causal-masked scaled scores -> attnRaw (scratch = attn output region),
// online per-row max (m) and sum-of-exp (l) -> Mrow/Lrow.
// ---------------------------------------------------------------------------
__global__ __launch_bounds__(32) void attn_scores_kernel(
    const float* __restrict__ Qp, const float* __restrict__ Kp,
    float* __restrict__ attnRaw, float* __restrict__ Mrow,
    float* __restrict__ Lrow, int S, int H, int nh, int dv)
{
  const int qt = blockIdx.x, h = blockIdx.y, b = blockIdx.z;
  const int lane = threadIdx.x & 31;
  const int half = lane >> 4;
  const int l16  = lane & 15;
  const int q0   = qt * 16;
  const float scale = 0.125f;  // 1/sqrt(dv=64)

  const size_t hb = ((size_t)b * S) * H + (size_t)h * dv;

  v16h aq[2];
#pragma unroll
  for (int kk = 0; kk < 2; ++kk)
#pragma unroll
    for (int e = 0; e < 16; ++e) {
      const int d = kk * 32 + kmap(e, half);
      aq[kk][e] = (_Float16)Qp[hb + (size_t)(q0 + l16) * H + d];
    }

  float m[8], l[8];
#pragma unroll
  for (int r = 0; r < 8; ++r) { m[r] = -__builtin_inff(); l[r] = 0.f; }

  float* rawBase = attnRaw + ((size_t)(b * nh + h) * S) * S;

  for (int kt = 0; kt <= qt; ++kt) {
    const int k0 = kt * 16;
    v16h bk[2];
#pragma unroll
    for (int kk = 0; kk < 2; ++kk)
#pragma unroll
      for (int e = 0; e < 16; ++e) {
        const int d = kk * 32 + kmap(e, half);
        bk[kk][e] = (_Float16)Kp[hb + (size_t)(k0 + l16) * H + d];
      }
    v8f c = {};
    c = WMMA_F32_F16(aq[0], bk[0], c);
    c = WMMA_F32_F16(aq[1], bk[1], c);

#pragma unroll
    for (int r = 0; r < 8; ++r) {
      const int qrow = q0 + r + 8 * half;
      const int key  = k0 + l16;
      float s = c[r] * scale;
      if (key > qrow) s = -__builtin_inff();      // causal
      rawBase[(size_t)qrow * S + key] = s;
      float tmax = s;
#pragma unroll
      for (int off = 8; off >= 1; off >>= 1)
        tmax = fmaxf(tmax, __shfl_xor(tmax, off, 32));
      const float mnew = fmaxf(m[r], tmax);
      float tsum = __expf(s - mnew);              // -inf -> 0
#pragma unroll
      for (int off = 8; off >= 1; off >>= 1)
        tsum += __shfl_xor(tsum, off, 32);
      l[r] = l[r] * __expf(m[r] - mnew) + tsum;
      m[r] = mnew;
    }
  }

  if (l16 == 0) {
    const size_t sb = (size_t)(b * nh + h) * S + q0 + 8 * half;
#pragma unroll
    for (int r = 0; r < 8; ++r) { Mrow[sb + r] = m[r]; Lrow[sb + r] = l[r]; }
  }
}

// ---------------------------------------------------------------------------
// attn_av: normalize raw scores -> attn probs (in place in d_out), and
// out = attn @ V (4 wmma per 32-key tile) -> concat buffer. Zero upper tri.
// ---------------------------------------------------------------------------
__global__ __launch_bounds__(32) void attn_av_kernel(
    const float* __restrict__ Vp, float* __restrict__ attn,
    const float* __restrict__ Mrow, const float* __restrict__ Lrow,
    float* __restrict__ Cc, int S, int H, int nh, int dv)
{
  const int qt = blockIdx.x, h = blockIdx.y, b = blockIdx.z;
  const int lane = threadIdx.x & 31;
  const int half = lane >> 4;
  const int l16  = lane & 15;
  const int q0   = qt * 16;
  const int myrow = q0 + l16;

  const size_t statb = (size_t)(b * nh + h) * S;
  const float mrow = Mrow[statb + myrow];
  const float linv = 1.0f / Lrow[statb + myrow];
  float* arow = attn + statb * S;

  const int kwrit = q0 + 16;
  const int ntile = (qt + 2) >> 1;

  const size_t vb = ((size_t)b * S) * H + (size_t)h * dv;

  v8f c[4];
#pragma unroll
  for (int j = 0; j < 4; ++j)
#pragma unroll
    for (int r = 0; r < 8; ++r) c[j][r] = 0.f;

  for (int kt = 0; kt < ntile; ++kt) {
    const int kb = kt * 32;
    v16h ap;
#pragma unroll
    for (int e = 0; e < 16; ++e) {
      const int key = kb + kmap(e, half);
      float p = 0.f;
      if (key < kwrit) {
        const float sraw = arow[(size_t)myrow * S + key];
        p = __expf(sraw - mrow) * linv;
      }
      ap[e] = (_Float16)p;
      arow[(size_t)myrow * S + key] = p;
    }
    v16h bv[4];
#pragma unroll
    for (int j = 0; j < 4; ++j)
#pragma unroll
      for (int e = 0; e < 16; ++e) {
        const int k = kb + kmap(e, half);
        const int d = j * 16 + l16;
        bv[j][e] = (_Float16)Vp[vb + (size_t)k * H + d];
      }
#pragma unroll
    for (int j = 0; j < 4; ++j)
      c[j] = WMMA_F32_F16(ap, bv[j], c[j]);
  }

  for (int r = 0; r < 16; ++r) {
    const size_t rb = (size_t)(q0 + r) * S;
    for (int col = ntile * 32 + lane; col < S; col += 32)
      arow[rb + col] = 0.f;
  }

#pragma unroll
  for (int j = 0; j < 4; ++j)
#pragma unroll
    for (int r = 0; r < 8; ++r) {
      const int row = q0 + r + 8 * half;
      const int col = h * dv + j * 16 + l16;
      Cc[((size_t)b * S + row) * H + col] = c[j][r];
    }
}

// ---------------------------------------------------------------------------
extern "C" void kernel_launch(void* const* d_in, const int* in_sizes, int n_in,
                              void* d_out, int out_size, void* d_ws, size_t ws_size,
                              hipStream_t stream) {
  const int B = 2, S = 2048, H = 1024, NH = 16, DV = 64;
  const int M = B * S;  // 4096

  const float* query = (const float*)d_in[0];
  const float* keyi  = (const float*)d_in[1];
  const float* value = (const float*)d_in[2];
  const float* Wq = (const float*)d_in[3];
  const float* bq = (const float*)d_in[4];
  const float* Wk = (const float*)d_in[5];
  const float* bk = (const float*)d_in[6];
  const float* Wv = (const float*)d_in[7];
  const float* bv = (const float*)d_in[8];
  const float* Wo = (const float*)d_in[9];
  const float* bo = (const float*)d_in[10];

  float* out0 = (float*)d_out;                // [B,S,H]
  float* attn = out0 + (size_t)B * S * H;     // [B,NH,S,S]

  float* ws = (float*)d_ws;
  const size_t MH = (size_t)M * H;
  float* Qp   = ws;
  float* Kp   = Qp + MH;
  float* Vp   = Kp + MH;
  float* Cc   = Vp + MH;
  float* Mrow = Cc + MH;
  float* Lrow = Mrow + (size_t)B * NH * S;

  dim3 ggrid(NDIM / PT_N, MDIM / PT_M);  // (16, 32)
  proj_gemm_kernel<<<ggrid, 256, 0, stream>>>(query, Wq, bq, Qp);
  proj_gemm_kernel<<<ggrid, 256, 0, stream>>>(keyi,  Wk, bk, Kp);
  proj_gemm_kernel<<<ggrid, 256, 0, stream>>>(value, Wv, bv, Vp);

  dim3 agrid(S / 16, NH, B);             // (128, 16, 2)
  attn_scores_kernel<<<agrid, 32, 0, stream>>>(Qp, Kp, attn, Mrow, Lrow, S, H, NH, DV);
  attn_av_kernel<<<agrid, 32, 0, stream>>>(Vp, attn, Mrow, Lrow, Cc, S, H, NH, DV);

  proj_gemm_kernel<<<ggrid, 256, 0, stream>>>(Cc, Wo, bo, out0);
}
